// RelativeMultiHeadAttention_44461501448570
// MI455X (gfx1250) — matrix operand order repaired and checked by reading, not compiled, not run on hardware
//
#include <hip/hip_runtime.h>
#include <stdint.h>

typedef __bf16 bf16_t;
typedef __attribute__((ext_vector_type(16))) __bf16 v16bf;
typedef __attribute__((ext_vector_type(8)))  float  v8f;

#define D_MODEL  1024
#define N_HEADS  16
#define HEAD_DIM 64
#define SEQ_LEN  2048
#define BATCH    2
#define M_TOTAL  (BATCH * SEQ_LEN)   // 4096
#define QKV_N    (3 * D_MODEL)       // 3072

__device__ __forceinline__ uint32_t f32_to_bf16(float f) {
  union { float f; uint32_t u; } v; v.f = f;
  uint32_t u = v.u;
  return (u + 0x7FFFu + ((u >> 16) & 1u)) >> 16;
}

__device__ __forceinline__ uint32_t pack_bf16x2(float lo, float hi) {
  return f32_to_bf16(lo) | (f32_to_bf16(hi) << 16);
}

// ---- CDNA5 async global->LDS staging (ASYNCcnt path) ----------------------
__device__ __forceinline__ void async_load_b128_to_lds(const void* gptr,
                                                       uint32_t lds_byte_addr) {
  asm volatile("global_load_async_to_lds_b128 %0, %1, off"
               :: "v"(lds_byte_addr), "v"((uint64_t)(uintptr_t)gptr)
               : "memory");
}

__device__ __forceinline__ void wait_asynccnt0() {
#if __has_builtin(__builtin_amdgcn_s_wait_asynccnt)
  __builtin_amdgcn_s_wait_asynccnt(0);
#else
  asm volatile("s_wait_asynccnt 0x0" ::: "memory");
#endif
}

// A-operand (16x32 bf16, M x K): lane holds row = lane%16,
// elements 0..7 -> K = base0..+7, elements 8..15 -> K = base0+16..+23,
// base0 = (lane>=16)*8.  Source row-major [row][k], leading dim ld.
__device__ __forceinline__ v16bf wmma_load_A(const uint16_t* base, int ld) {
  const int lane = threadIdx.x & 31;
  const int row  = lane & 15;
  const int k0   = (lane >> 4) << 3;
  const bf16_t* p0 = (const bf16_t*)(base + row * ld + k0);
  const bf16_t* p1 = p0 + 16;
  v16bf a;
#pragma unroll
  for (int i = 0; i < 8; ++i) { a[i] = p0[i]; a[i + 8] = p1[i]; }
  return a;
}

// B-operand (32x16 bf16, K x N): lane holds col n = lane%16,
// elements 0..7 -> K = base0..+7, elements 8..15 -> K = base0+16..+23,
// base0 = (lane>=16)*8.  Source stored [n][k] (ld = K stride).
__device__ __forceinline__ v16bf wmma_load_B(const uint16_t* base, int ld) {
  const int lane = threadIdx.x & 31;
  const int n    = lane & 15;
  const int k0   = (lane >> 4) << 3;
  const bf16_t* p = (const bf16_t*)(base + n * ld + k0);
  v16bf b;
#pragma unroll
  for (int i = 0; i < 8; ++i) { b[i] = p[i]; b[i + 8] = p[i + 16]; }
  return b;
}

__device__ __forceinline__ v8f wmma_bf16(v16bf a, v16bf b, v8f c) {
  return __builtin_amdgcn_wmma_f32_16x16x32_bf16(false, a, false, b,
                                                 (short)0, c, false, false);
}

// ---------------------------------------------------------------------------
// Kernel 1: fused QKV projection  (x[4096,1024] @ qkv_w[1024,3072] + b)
// writes bf16 Q (scaled by dk^-0.5), K as (B,H,L,dk), V^T as (B,H,dk,L)
// ---------------------------------------------------------------------------
#define G_TM 128
#define G_TN 64
#define G_TK 32

__global__ __launch_bounds__(256)
void qkv_gemm_kernel(const float* __restrict__ x, const float* __restrict__ w,
                     const float* __restrict__ bias,
                     uint16_t* __restrict__ Qb, uint16_t* __restrict__ Kb,
                     uint16_t* __restrict__ Vt) {
  __shared__ __attribute__((aligned(16))) uint16_t As[G_TM * G_TK];
  __shared__ __attribute__((aligned(16))) uint16_t Ws[G_TN * G_TK];
  const int ntile = blockIdx.x * G_TN;
  const int mtile = blockIdx.y * G_TM;
  const int t     = threadIdx.x;
  const int wave  = t >> 5;
  const int lane  = t & 31;

  v8f acc[4];
#pragma unroll
  for (int j = 0; j < 4; ++j) acc[j] = v8f{};

  for (int k0 = 0; k0 < D_MODEL; k0 += G_TK) {
    { // stage A tile 128x32 (f32 -> bf16, packed 32-bit LDS stores)
      const int row = t >> 1;
      const int cb  = (t & 1) * 16;
      const float* src = x + (size_t)(mtile + row) * D_MODEL + k0 + cb;
      if (k0 + G_TK < D_MODEL) __builtin_prefetch(src + G_TK, 0, 0);
      uint32_t* dst = (uint32_t*)&As[row * G_TK + cb];
#pragma unroll
      for (int i = 0; i < 8; ++i) dst[i] = pack_bf16x2(src[2 * i], src[2 * i + 1]);
    }
    { // stage W tile transposed: Ws[n][k], 64 x 32
      const int kk = t >> 3;
      const int nb = (t & 7) * 8;
      const float* src = w + (size_t)(k0 + kk) * QKV_N + ntile + nb;
#pragma unroll
      for (int i = 0; i < 8; ++i) Ws[(nb + i) * G_TK + kk] = (uint16_t)f32_to_bf16(src[i]);
    }
    __syncthreads();
    v16bf a = wmma_load_A(&As[wave * 16 * G_TK], G_TK);
#pragma unroll
    for (int j = 0; j < 4; ++j) {
      v16bf b = wmma_load_B(&Ws[j * 16 * G_TK], G_TK);
      acc[j] = wmma_bf16(a, b, acc[j]);
    }
    __syncthreads();
  }

  // epilogue: +bias, scatter to Q / K / V^T (bf16)
  const int rhi  = (lane >> 4) * 8;
  const int ncol = lane & 15;
#pragma unroll
  for (int j = 0; j < 4; ++j) {
    const int n = ntile + j * 16 + ncol;
    const int s = n >> 10;          // 0=q 1=k 2=v
    const int h = (n >> 6) & 15;
    const int d = n & 63;
    const float bv = bias[n];
#pragma unroll
    for (int r = 0; r < 8; ++r) {
      const int m    = mtile + wave * 16 + rhi + r;
      const int bidx = m >> 11;
      const int lq   = m & 2047;
      const float v  = acc[j][r] + bv;
      const size_t hb = (size_t)(bidx * N_HEADS + h);
      if (s == 0)      Qb[(hb * SEQ_LEN + lq) * HEAD_DIM + d] = (uint16_t)f32_to_bf16(v * 0.125f);
      else if (s == 1) Kb[(hb * SEQ_LEN + lq) * HEAD_DIM + d] = (uint16_t)f32_to_bf16(v);
      else             Vt[(hb * HEAD_DIM + d) * SEQ_LEN + lq] = (uint16_t)f32_to_bf16(v);
    }
  }
}

// ---------------------------------------------------------------------------
// Kernel 2: flash attention, 4 waves per block; waves share one async-staged
// K/V chunk, each wave owns a private 16-query tile.
// ---------------------------------------------------------------------------
#define AW 4

__global__ __launch_bounds__(32 * AW)
void attn_kernel(const uint16_t* __restrict__ Qb, const uint16_t* __restrict__ Kb,
                 const uint16_t* __restrict__ Vt, const float* __restrict__ bias_table,
                 float* __restrict__ Obuf) {
  __shared__ __attribute__((aligned(16))) uint16_t Ks[32 * 64];       // [key][d]
  __shared__ __attribute__((aligned(16))) uint16_t Vs[64 * 32];       // [d][key]
  __shared__ __attribute__((aligned(16))) float    Sld[AW][16 * 32];
  __shared__ __attribute__((aligned(16))) uint16_t Pld[AW][16 * 32];
  __shared__ float biasrow[2112];
  __shared__ float rowscale[AW][16];

  const int qbase0 = blockIdx.x * (16 * AW);
  const int h      = blockIdx.y;
  const int b      = blockIdx.z;
  const int tid    = threadIdx.x;
  const int wq     = tid >> 5;
  const int lane   = tid & 31;

  // stage rel-bias row for the 64-query union; lookup index = k - q_local + 63
  for (int tt = tid; tt < 2111; tt += 32 * AW) {
    biasrow[tt] = bias_table[(size_t)(tt + 1984 - qbase0) * N_HEADS + h];
  }

  const size_t hb = (size_t)(b * N_HEADS + h);
  const uint16_t* qptr = Qb + (hb * SEQ_LEN + qbase0 + wq * 16) * HEAD_DIM;
  const v16bf aq0 = wmma_load_A(qptr, HEAD_DIM);       // d 0..31
  const v16bf aq1 = wmma_load_A(qptr + 32, HEAD_DIM);  // d 32..63

  v8f o[4];
#pragma unroll
  for (int j = 0; j < 4; ++j) o[j] = v8f{};
  float m_run = -3.0e38f, s_run = 0.f;

  const uint16_t* kptr = Kb + hb * SEQ_LEN * HEAD_DIM;
  const uint16_t* vptr = Vt + hb * HEAD_DIM * SEQ_LEN;
  const uint32_t ks_lds = (uint32_t)(uintptr_t)Ks;
  const uint32_t vs_lds = (uint32_t)(uintptr_t)Vs;
  __syncthreads();

  for (int kc = 0; kc < SEQ_LEN; kc += 32) {
    // async stage K chunk: 32 keys x 128B = 256 16B chunks over 128 lanes
#pragma unroll
    for (int i = 0; i < 2; ++i) {
      const int c = tid + i * 128;
      async_load_b128_to_lds(
          kptr + (size_t)(kc + (c >> 3)) * HEAD_DIM + (c & 7) * 8,
          ks_lds + c * 16);
    }
    // async stage V chunk: 64 d-rows x 64B = 256 16B chunks
#pragma unroll
    for (int i = 0; i < 2; ++i) {
      const int c = tid + i * 128;
      async_load_b128_to_lds(
          vptr + (size_t)(c >> 2) * SEQ_LEN + kc + (c & 3) * 8,
          vs_lds + c * 16);
    }
    wait_asynccnt0();
    __syncthreads();

    // S = Qs K^T + bias  (16 q x 32 k per wave)
#pragma unroll
    for (int n = 0; n < 2; ++n) {
      v8f s = v8f{};
      s = wmma_bf16(aq0, wmma_load_B(&Ks[(n * 16) * 64], 64), s);
      s = wmma_bf16(aq1, wmma_load_B(&Ks[(n * 16) * 64 + 32], 64), s);
      const int col = n * 16 + (lane & 15);
      const int kg  = kc + col;
      const int rhi = (lane >> 4) * 8;
#pragma unroll
      for (int r = 0; r < 8; ++r) {
        const int row = rhi + r;
        Sld[wq][row * 32 + col] = s[r] + biasrow[kg - (wq * 16 + row) + 63];
      }
    }
    __syncthreads();

    // online softmax: lanes 0..15 of each wave own one query row each
    if (lane < 16) {
      float mx = m_run;
#pragma unroll
      for (int j2 = 0; j2 < 32; ++j2) mx = fmaxf(mx, Sld[wq][lane * 32 + j2]);
      const float corr = __expf(m_run - mx);
      float ssum = s_run * corr;
      uint32_t* p32 = (uint32_t*)&Pld[wq][lane * 32];
#pragma unroll
      for (int j2 = 0; j2 < 32; j2 += 2) {
        const float p0 = __expf(Sld[wq][lane * 32 + j2] - mx);
        const float p1 = __expf(Sld[wq][lane * 32 + j2 + 1] - mx);
        ssum += p0 + p1;
        p32[j2 >> 1] = pack_bf16x2(p0, p1);
      }
      rowscale[wq][lane] = corr;
      m_run = mx; s_run = ssum;
    }
    __syncthreads();

    { // rescale accumulators by exp(m_old - m_new)
      const int rhi = (lane >> 4) * 8;
#pragma unroll
      for (int r = 0; r < 8; ++r) {
        const float c = rowscale[wq][rhi + r];
#pragma unroll
        for (int j2 = 0; j2 < 4; ++j2) o[j2][r] *= c;
      }
    }
    // O += P V : A = P(16x32), B tiles from Vs[d][key]
    const v16bf ap = wmma_load_A(Pld[wq], 32);
#pragma unroll
    for (int dt = 0; dt < 4; ++dt) {
      v16bf bv = wmma_load_B(&Vs[(dt * 16) * 32], 32);
      o[dt] = wmma_bf16(ap, bv, o[dt]);
    }
    __syncthreads();
  }

  if (lane < 16) rowscale[wq][lane] = 1.0f / s_run;
  __syncthreads();

  const int rhi  = (lane >> 4) * 8;
  const int ncol = lane & 15;
#pragma unroll
  for (int dt = 0; dt < 4; ++dt) {
    const int d = dt * 16 + ncol;
#pragma unroll
    for (int r = 0; r < 8; ++r) {
      const int row = rhi + r;
      Obuf[((size_t)(b * SEQ_LEN) + qbase0 + wq * 16 + row) * D_MODEL + h * HEAD_DIM + d] =
          o[dt][r] * rowscale[wq][row];
    }
  }
}

// ---------------------------------------------------------------------------
// Kernel 3: output projection  (O[4096,1024] @ proj_w[1024,1024] + b -> out)
// ---------------------------------------------------------------------------
__global__ __launch_bounds__(256)
void proj_gemm_kernel(const float* __restrict__ A, const float* __restrict__ w,
                      const float* __restrict__ bias, float* __restrict__ out) {
  __shared__ __attribute__((aligned(16))) uint16_t As[G_TM * G_TK];
  __shared__ __attribute__((aligned(16))) uint16_t Ws[G_TN * G_TK];
  const int ntile = blockIdx.x * G_TN;
  const int mtile = blockIdx.y * G_TM;
  const int t     = threadIdx.x;
  const int wave  = t >> 5;
  const int lane  = t & 31;

  v8f acc[4];
#pragma unroll
  for (int j = 0; j < 4; ++j) acc[j] = v8f{};

  for (int k0 = 0; k0 < D_MODEL; k0 += G_TK) {
    {
      const int row = t >> 1;
      const int cb  = (t & 1) * 16;
      const float* src = A + (size_t)(mtile + row) * D_MODEL + k0 + cb;
      if (k0 + G_TK < D_MODEL) __builtin_prefetch(src + G_TK, 0, 0);
      uint32_t* dst = (uint32_t*)&As[row * G_TK + cb];
#pragma unroll
      for (int i = 0; i < 8; ++i) dst[i] = pack_bf16x2(src[2 * i], src[2 * i + 1]);
    }
    {
      const int kk = t >> 3;
      const int nb = (t & 7) * 8;
      const float* src = w + (size_t)(k0 + kk) * D_MODEL + ntile + nb;
#pragma unroll
      for (int i = 0; i < 8; ++i) Ws[(nb + i) * G_TK + kk] = (uint16_t)f32_to_bf16(src[i]);
    }
    __syncthreads();
    v16bf a = wmma_load_A(&As[wave * 16 * G_TK], G_TK);
#pragma unroll
    for (int j = 0; j < 4; ++j) {
      v16bf b = wmma_load_B(&Ws[j * 16 * G_TK], G_TK);
      acc[j] = wmma_bf16(a, b, acc[j]);
    }
    __syncthreads();
  }

  const int rhi  = (lane >> 4) * 8;
  const int ncol = lane & 15;
#pragma unroll
  for (int j = 0; j < 4; ++j) {
    const int n = ntile + j * 16 + ncol;
    const float bv = bias[n];
#pragma unroll
    for (int r = 0; r < 8; ++r) {
      const int m = mtile + wave * 16 + rhi + r;
      out[(size_t)m * D_MODEL + n] = acc[j][r] + bv;
    }
  }
}

// ---------------------------------------------------------------------------
extern "C" void kernel_launch(void* const* d_in, const int* in_sizes, int n_in,
                              void* d_out, int out_size, void* d_ws, size_t ws_size,
                              hipStream_t stream) {
  const float* x          = (const float*)d_in[0];
  const float* qkv_w      = (const float*)d_in[1];
  const float* qkv_b      = (const float*)d_in[2];
  const float* proj_w     = (const float*)d_in[3];
  const float* proj_b     = (const float*)d_in[4];
  const float* bias_table = (const float*)d_in[5];
  float* out = (float*)d_out;

  const size_t hsz = (size_t)BATCH * N_HEADS * SEQ_LEN * HEAD_DIM; // 4M elems
  uint16_t* Qb = (uint16_t*)d_ws;
  uint16_t* Kb = Qb + hsz;
  uint16_t* Vt = Kb + hsz;
  float* Obuf  = (float*)(Vt + hsz);

  dim3 gq(QKV_N / G_TN, M_TOTAL / G_TM);        // (48, 32)
  qkv_gemm_kernel<<<gq, 256, 0, stream>>>(x, qkv_w, qkv_b, Qb, Kb, Vt);

  dim3 ga(SEQ_LEN / (16 * AW), N_HEADS, BATCH); // (32, 16, 2)
  attn_kernel<<<ga, 32 * AW, 0, stream>>>(Qb, Kb, Vt, bias_table, Obuf);

  dim3 gp(D_MODEL / G_TN, M_TOTAL / G_TM);      // (16, 32)
  proj_gemm_kernel<<<gp, 256, 0, stream>>>(Obuf, proj_w, proj_b, out);
}
